// DecoupledMVRowSelfAttnProcessor2_0_57690000720126
// MI455X (gfx1250) — matrix operand order repaired and checked by reading, not compiled, not run on hardware
//
#include <hip/hip_runtime.h>
#include <stdint.h>

// ---------------------------------------------------------------------------
// Types for CDNA5 WMMA (wave32): v_wmma_f32_16x16x32_bf16
// ---------------------------------------------------------------------------
typedef __bf16 bf16;
typedef __attribute__((ext_vector_type(8)))  bf16  bf16x8;
typedef __attribute__((ext_vector_type(16))) bf16  v16bf;
typedef __attribute__((ext_vector_type(8)))  float v8f;
typedef __attribute__((ext_vector_type(4)))  unsigned int u32x4;
typedef __attribute__((ext_vector_type(8)))  int i32x8;
typedef __attribute__((ext_vector_type(4)))  int i32x4;

__device__ __forceinline__ bf16 f2bf(float f) {
  union { float f; uint32_t u; } x; x.f = f;
  uint32_t r = x.u + 0x7fffu + ((x.u >> 16) & 1u);   // round-to-nearest-even
  uint16_t h = (uint16_t)(r >> 16);
  return __builtin_bit_cast(bf16, h);
}

__device__ __forceinline__ v8f wmma_bf16(v16bf a, v16bf b, v8f c) {
  // D = A(16x32 bf16) * B(32x16 bf16) + C(16x16 f32)
  return __builtin_amdgcn_wmma_f32_16x16x32_bf16(
      /*neg_a=*/false, a, /*neg_b=*/false, b,
      /*c_mod=*/(short)0, c, /*reuse_a=*/false, /*reuse_b=*/false);
}

__device__ __forceinline__ v16bf cat8(bf16x8 lo, bf16x8 hi) {
  return __builtin_shufflevector(lo, hi, 0,1,2,3,4,5,6,7,8,9,10,11,12,13,14,15);
}

// A-fragment (16x32 bf16) load: row-major source, per-lane layout per ISA 7.12.2:
// lanes 0-15: M=lane, K = 0..7 and 16..23 ; lanes 16-31: M=lane-16, K = 8..15, 24..31
__device__ __forceinline__ v16bf load_a_frag(const bf16* base, int row_stride,
                                             int lane, int kofs) {
  int row  = lane & 15;
  int kh   = (lane >> 4) * 8;
  const bf16* p = base + (size_t)row * row_stride + kofs + kh;
  bf16x8 lo = *(const bf16x8*)(p);        // K = kh..kh+7
  bf16x8 hi = *(const bf16x8*)(p + 16);   // K = 16+kh..16+kh+7
  return cat8(lo, hi);
}

// ---------------------------------------------------------------------------
// Tensor Data Mover: 2D tile (bf16) global -> LDS, descriptor per D# spec.
// group0: [1:0]count=1 | lds_addr[63:32] | global_addr[120:64] | type[127:126]=2
// group1: data_size=1(2B) @[17:16], tensor_dim0[79:48], tensor_dim1[111:80],
//         tile_dim0[127:112], tile_dim1[143:128], tensor_dim0_stride[207:160]
// This toolchain exposes the 6-arg builtin (g0,g1,g2,g3,g4,cpol).
// ---------------------------------------------------------------------------
__device__ __forceinline__ void tdm_load_tile_2d(
    uint64_t gaddr_bytes, uint32_t lds_off_bytes,
    uint32_t tensor_d0, uint32_t tensor_d1, uint32_t stride0_elems,
    uint32_t tile_d0, uint32_t tile_d1) {
  u32x4 g0;
  g0[0] = 1u;                                            // count=1 valid descr
  g0[1] = lds_off_bytes;                                 // lds_addr
  g0[2] = (uint32_t)(gaddr_bytes & 0xffffffffu);         // global_addr[31:0]
  g0[3] = (uint32_t)((gaddr_bytes >> 32) & 0x01ffffffu)  // global_addr[56:32]
          | (2u << 30);                                  // type = 2 ("image")
  i32x8 g1;
  g1[0] = (int)(1u << 16);                               // data_size = 2 bytes
  g1[1] = (int)((tensor_d0 & 0xffffu) << 16);            // td0[15:0] -> bits 63:48
  g1[2] = (int)((tensor_d0 >> 16) | ((tensor_d1 & 0xffffu) << 16));
  g1[3] = (int)((tensor_d1 >> 16) | (tile_d0 << 16));    // tile_dim0
  g1[4] = (int)(tile_d1 & 0xffffu);                      // tile_dim1 (tile_dim2=0)
  g1[5] = (int)stride0_elems;                            // dim0 stride low 32
  g1[6] = 0;                                             // stride hi / dim1 stride
  g1[7] = 0;
  i32x4 gz  = {0, 0, 0, 0};                              // groups 2/3 unused (2D)
  i32x8 gz8 = {0, 0, 0, 0, 0, 0, 0, 0};                  // group 4 unused
  __builtin_amdgcn_tensor_load_to_lds(g0, g1, gz, gz, gz8, 0);
}

// ---------------------------------------------------------------------------
// Elementwise helpers
// ---------------------------------------------------------------------------
__global__ void cvt_f32_bf16_kernel(const float* __restrict__ in,
                                    bf16* __restrict__ out, int n) {
  int i = blockIdx.x * 256 + threadIdx.x;
  if (i < n) out[i] = f2bf(in[i]);
}

__global__ void copy_f32_kernel(const float* __restrict__ in,
                                float* __restrict__ out, int n) {
  int i = blockIdx.x * 256 + threadIdx.x;
  if (i < n) out[i] = in[i];
}

// K [B,S,C] (head cols) -> Kt [B*h][64][1024]  (scores B-fragment is contiguous)
__global__ void transpose_k_kernel(const bf16* __restrict__ k,
                                   bf16* __restrict__ kt) {
  int idx = blockIdx.x * 256 + threadIdx.x;          // 12*16*64*1024 total
  if (idx >= 12 * 16 * 64 * 1024) return;
  int s  = idx & 1023;
  int d  = (idx >> 10) & 63;
  int bh = idx >> 16;
  int head = bh & 15, batch = bh >> 4;
  kt[idx] = k[((size_t)batch * 1024 + s) * 1024 + head * 64 + d];
}

// K_mv [B,S,C] -> Ktg [b][ih][h][64][192]   (row bank gathered across views)
__global__ void gather_mv_kt_kernel(const bf16* __restrict__ k,
                                    bf16* __restrict__ ktg) {
  int idx = blockIdx.x * 256 + threadIdx.x;          // 64*16*64*192 total
  if (idx >= 64 * 16 * 64 * 192) return;
  int j = idx % 192;       int t = idx / 192;
  int d = t & 63;          t >>= 6;
  int h = t & 15;          t >>= 4;
  int ihrow = t & 31;      int b_ = t >> 5;
  int batch = b_ * 6 + (j >> 5);
  int s     = ihrow * 32 + (j & 31);
  ktg[idx] = k[((size_t)batch * 1024 + s) * 1024 + h * 64 + d];
}

// V_mv [B,S,C] -> Vg [b][ih][192][1024]
__global__ void gather_mv_v_kernel(const bf16* __restrict__ v,
                                   bf16* __restrict__ vg) {
  int idx = blockIdx.x * 256 + threadIdx.x;          // 64*192*1024 total
  if (idx >= 64 * 192 * 1024) return;
  int c = idx & 1023;      int t = idx >> 10;
  int j = t % 192;         t /= 192;
  int ihrow = t & 31;      int b_ = t >> 5;
  int batch = b_ * 6 + (j >> 5);
  int s     = ihrow * 32 + (j & 31);
  vg[idx] = v[((size_t)batch * 1024 + s) * 1024 + c];
}

// ---------------------------------------------------------------------------
// Tiled bf16 WMMA GEMM with TDM-staged, double-buffered LDS tiles.
// Y[M=12288,1024] = A[.,1024] @ W[1024,1024]
// Block tile 128x128, BK=32, 256 threads (8 waves), wave tile 32x64.
// Wave 0 drives the Tensor Data Mover; all waves run fragment loads + WMMA.
// MODE 0: store bf16     MODE 1: d_out += Y + bias  (fp32 accumulate)
// ---------------------------------------------------------------------------
template <int MODE>
__global__ __launch_bounds__(256)
void gemm_bf16_kernel(const bf16* __restrict__ A, const bf16* __restrict__ W,
                      bf16* __restrict__ Ybf, float* __restrict__ Yacc,
                      const float* __restrict__ bias) {
  // [buf][ A tile 128x32 (4096) | B tile 32x128 (4096) ]   32 KB total
  __shared__ bf16 smem[2][8192];
  int t = threadIdx.x, lane = t & 31, wave = t >> 5;
  int wm = wave & 3, wn = wave >> 2;            // wave grid 4x2
  int bm = blockIdx.x * 128, bn = blockIdx.y * 128;
  int col = lane & 15, half = lane >> 4;

  const uint64_t Ab = (uint64_t)(uintptr_t)A;
  const uint64_t Wb = (uint64_t)(uintptr_t)W;

  v8f acc[2][4];
#pragma unroll
  for (int mt = 0; mt < 2; ++mt)
#pragma unroll
    for (int nt = 0; nt < 4; ++nt) acc[mt][nt] = (v8f){};

  // prologue: DMA first tile pair into buffer 0
  if (wave == 0) {
    tdm_load_tile_2d(Ab + ((uint64_t)bm * 1024) * 2, 0u,
                     1024, 12288, 1024, /*tile*/ 32, 128);
    tdm_load_tile_2d(Wb + ((uint64_t)bn) * 2, 16384u,
                     1024, 1024, 1024, /*tile*/ 128, 32);
    __builtin_amdgcn_s_wait_tensorcnt(0);
  }
  __syncthreads();

  for (int k0 = 0; k0 < 1024; k0 += 32) {
    int cur = (k0 >> 5) & 1;
    // prefetch next tile pair into the other buffer while we compute
    if (wave == 0 && k0 + 32 < 1024) {
      uint32_t nb = (cur ^ 1) * 16384u;
      tdm_load_tile_2d(Ab + ((uint64_t)bm * 1024 + (k0 + 32)) * 2, nb,
                       1024, 12288, 1024, 32, 128);
      tdm_load_tile_2d(Wb + ((uint64_t)(k0 + 32) * 1024 + bn) * 2, nb + 8192u,
                       1024, 1024, 1024, 128, 32);
    }

    const bf16* sA = &smem[cur][0];
    const bf16* sB = &smem[cur][4096];
    v16bf af[2], bfr[4];
#pragma unroll
    for (int mt = 0; mt < 2; ++mt)
      af[mt] = load_a_frag(sA + (wm * 32 + mt * 16) * 32, 32, lane, 0);
#pragma unroll
    for (int nt = 0; nt < 4; ++nt)   // B lane layout: N=lane%16, K in table order
      for (int s = 0; s < 16; ++s)
        bfr[nt][s] = sB[((s & 7) + half * 8 + (s >> 3) * 16) * 128 + wn * 64 + nt * 16 + col];

#pragma unroll
    for (int mt = 0; mt < 2; ++mt)
#pragma unroll
      for (int nt = 0; nt < 4; ++nt)
        acc[mt][nt] = wmma_bf16(af[mt], bfr[nt], acc[mt][nt]);

    if (wave == 0) __builtin_amdgcn_s_wait_tensorcnt(0);
    __syncthreads();
  }

#pragma unroll
  for (int mt = 0; mt < 2; ++mt) {
#pragma unroll
    for (int nt = 0; nt < 4; ++nt) {
#pragma unroll
      for (int r = 0; r < 8; ++r) {
        int m = bm + wm * 32 + mt * 16 + r + half * 8;
        int n = bn + wn * 64 + nt * 16 + col;
        size_t idx = (size_t)m * 1024 + n;
        float val = acc[mt][nt][r];
        if (MODE == 0) Ybf[idx] = f2bf(val);
        else           Yacc[idx] += val + bias[n];
      }
    }
  }
}

// ---------------------------------------------------------------------------
// Per-wave flash attention: 16 query rows, d=64, online softmax.
// Q row-major [.,64-chunk], Kt [64][Sk] (contiguous keys), V [key][64-chunk].
// ---------------------------------------------------------------------------
__device__ __forceinline__ void flash_wave(
    const bf16* __restrict__ Qb, int q_rs,
    const bf16* __restrict__ Kt, int kt_rs,
    const bf16* __restrict__ Vb, int v_rs,
    bf16* __restrict__ Ob, int o_rs,
    int Sk, float* pst, int lane) {
  const float scale = 0.125f;                 // 1/sqrt(64)
  int col = lane & 15, half = lane >> 4;

  v16bf aq0 = load_a_frag(Qb, q_rs, lane, 0);
  v16bf aq1 = load_a_frag(Qb, q_rs, lane, 32);

  v8f o0 = (v8f){}, o1 = (v8f){}, o2 = (v8f){}, o3 = (v8f){};
  float m_run[8], l_run[8];
#pragma unroll
  for (int r = 0; r < 8; ++r) { m_run[r] = -1e30f; l_run[r] = 0.0f; }

  for (int kb = 0; kb < Sk; kb += 32) {
    // scores S = Q K^T (two 16-key tiles, contract over d in two 32-chunks)
    v16bf bk00, bk10, bk01, bk11;
    for (int s = 0; s < 16; ++s) {
      int kk = (s & 7) + half * 8 + (s >> 3) * 16;
      bk00[s] = Kt[(size_t)(kk)      * kt_rs + kb + col];
      bk10[s] = Kt[(size_t)(32 + kk) * kt_rs + kb + col];
      bk01[s] = Kt[(size_t)(kk)      * kt_rs + kb + 16 + col];
      bk11[s] = Kt[(size_t)(32 + kk) * kt_rs + kb + 16 + col];
    }
    v8f s0 = wmma_bf16(aq0, bk00, (v8f){});
    s0     = wmma_bf16(aq1, bk10, s0);
    v8f s1 = wmma_bf16(aq0, bk01, (v8f){});
    s1     = wmma_bf16(aq1, bk11, s1);

    // online softmax per row (row r in lanes 0-15 / row r+8 in lanes 16-31)
#pragma unroll
    for (int r = 0; r < 8; ++r) {
      float a = s0[r] * scale, b = s1[r] * scale;
      float mx = fmaxf(a, b);
#pragma unroll
      for (int msk = 8; msk >= 1; msk >>= 1)
        mx = fmaxf(mx, __shfl_xor(mx, msk, 32));
      float mnew = fmaxf(m_run[r], mx);
      float sc = __expf(m_run[r] - mnew);
      float p0 = __expf(a - mnew), p1 = __expf(b - mnew);
      float rs = p0 + p1;
#pragma unroll
      for (int msk = 8; msk >= 1; msk >>= 1)
        rs += __shfl_xor(rs, msk, 32);
      l_run[r] = l_run[r] * sc + rs;
      m_run[r] = mnew;
      o0[r] *= sc; o1[r] *= sc; o2[r] *= sc; o3[r] *= sc;
      int row = r + half * 8;
      pst[row * 32 + col]      = p0;         // stage P tile 16x32 in LDS
      pst[row * 32 + 16 + col] = p1;
    }
    asm volatile("s_wait_dscnt 0x0" ::: "memory");

    // re-fragment P (C layout -> A layout) as bf16
    v16bf ap;
    {
      int row = lane & 15, kh = half * 8;
      const float* pr = pst + row * 32;
      float4 q0 = *(const float4*)(pr + kh);
      float4 q1 = *(const float4*)(pr + kh + 4);
      float4 q2 = *(const float4*)(pr + 16 + kh);
      float4 q3 = *(const float4*)(pr + 16 + kh + 4);
      ap[0] = f2bf(q0.x); ap[1] = f2bf(q0.y); ap[2]  = f2bf(q0.z); ap[3]  = f2bf(q0.w);
      ap[4] = f2bf(q1.x); ap[5] = f2bf(q1.y); ap[6]  = f2bf(q1.z); ap[7]  = f2bf(q1.w);
      ap[8] = f2bf(q2.x); ap[9] = f2bf(q2.y); ap[10] = f2bf(q2.z); ap[11] = f2bf(q2.w);
      ap[12]= f2bf(q3.x); ap[13]= f2bf(q3.y); ap[14] = f2bf(q3.z); ap[15] = f2bf(q3.w);
    }

    // O += P @ V  (keys = contraction, 4 d-tiles of 16)
    v16bf bv0, bv1, bv2, bv3;
    for (int s = 0; s < 16; ++s) {
      int kk = (s & 7) + half * 8 + (s >> 3) * 16;
      const bf16* vrow = Vb + (size_t)(kb + kk) * v_rs + col;
      bv0[s] = vrow[0]; bv1[s] = vrow[16]; bv2[s] = vrow[32]; bv3[s] = vrow[48];
    }
    o0 = wmma_bf16(ap, bv0, o0);
    o1 = wmma_bf16(ap, bv1, o1);
    o2 = wmma_bf16(ap, bv2, o2);
    o3 = wmma_bf16(ap, bv3, o3);
  }

#pragma unroll
  for (int r = 0; r < 8; ++r) {
    float inv = 1.0f / l_run[r];
    bf16* orow = Ob + (size_t)(r + half * 8) * o_rs;
    orow[col]      = f2bf(o0[r] * inv);
    orow[16 + col] = f2bf(o1[r] * inv);
    orow[32 + col] = f2bf(o2[r] * inv);
    orow[48 + col] = f2bf(o3[r] * inv);
  }
}

// Dense SDPA: grid (8 qblocks, 16 heads, 12 batch); wave = 16 query rows
__global__ __launch_bounds__(256)
void attn_dense_kernel(const bf16* __restrict__ Q, const bf16* __restrict__ Kt,
                       const bf16* __restrict__ V, bf16* __restrict__ O) {
  __shared__ float pst[8 * 512];
  int lane = threadIdx.x & 31, wave = threadIdx.x >> 5;
  int head = blockIdx.y, batch = blockIdx.z;
  int q0 = blockIdx.x * 128 + wave * 16;
  const bf16* Qb  = Q  + ((size_t)batch * 1024 + q0) * 1024 + head * 64;
  const bf16* Ktb = Kt + (size_t)(batch * 16 + head) * 64 * 1024;
  const bf16* Vb  = V  + (size_t)batch * 1024 * 1024 + head * 64;
  bf16*       Ob  = O  + ((size_t)batch * 1024 + q0) * 1024 + head * 64;
  flash_wave(Qb, 1024, Ktb, 1024, Vb, 1024, Ob, 1024, 1024,
             pst + wave * 512, lane);
}

// Multi-view row SDPA: 384 (b,view,row) groups x 16 heads x 2 q-halves, Sk=192
__global__ __launch_bounds__(256)
void attn_mv_kernel(const bf16* __restrict__ Q, const bf16* __restrict__ Ktg,
                    const bf16* __restrict__ Vg, bf16* __restrict__ O) {
  __shared__ float pst[8 * 512];
  int lane = threadIdx.x & 31, wave = threadIdx.x >> 5;
  int w = blockIdx.x * 8 + wave;            // 12288 wave tasks
  int head  = w & 15;
  int qhalf = (w >> 4) & 1;
  int grp   = w >> 5;                       // 0..383
  int b_ = grp / 192, rem = grp % 192;
  int view = rem >> 5, ihrow = rem & 31;
  int batch = b_ * 6 + view;
  int s0 = ihrow * 32 + qhalf * 16;
  int bank = b_ * 32 + ihrow;
  const bf16* Qb  = Q   + ((size_t)batch * 1024 + s0) * 1024 + head * 64;
  const bf16* Ktb = Ktg + ((size_t)bank * 16 + head) * 64 * 192;
  const bf16* Vb  = Vg  + (size_t)bank * 192 * 1024 + head * 64;
  bf16*       Ob  = O   + ((size_t)batch * 1024 + s0) * 1024 + head * 64;
  flash_wave(Qb, 1024, Ktb, 192, Vb, 1024, Ob, 1024, 192,
             pst + wave * 512, lane);
}

// ---------------------------------------------------------------------------
// Host orchestration
// ---------------------------------------------------------------------------
extern "C" void kernel_launch(void* const* d_in, const int* in_sizes, int n_in,
                              void* d_out, int out_size, void* d_ws, size_t ws_size,
                              hipStream_t stream) {
  (void)in_sizes; (void)n_in; (void)out_size; (void)ws_size;
  const float* hs   = (const float*)d_in[0];
  const float* ref  = (const float*)d_in[1];
  // d_in order: Wq,Wk,Wv,Wout, Wq_mv,Wk_mv,Wv_mv,Wout_mv, Wq_ref,Wk_ref,Wv_ref,Wout_ref
  const float* bout     = (const float*)d_in[14];
  const float* bout_mv  = (const float*)d_in[15];
  const float* bout_ref = (const float*)d_in[16];
  float* out = (float*)d_out;

  const int NACT = 12 * 1024 * 1024;         // 12,582,912 elems per region
  const size_t RB = (size_t)NACT * sizeof(bf16);
  char* ws = (char*)d_ws;
  bf16* hs_bf  = (bf16*)(ws + 0 * RB);
  bf16* ref_bf = (bf16*)(ws + 1 * RB);
  bf16* w_bf   = (bf16*)(ws + 2 * RB);       // 12 x 1M weights
  bf16* q_bf   = (bf16*)(ws + 3 * RB);
  bf16* k_bf   = (bf16*)(ws + 4 * RB);
  bf16* v_bf   = (bf16*)(ws + 5 * RB);
  bf16* kt_bf  = (bf16*)(ws + 6 * RB);       // also reused as mv key bank
  bf16* g_bf   = (bf16*)(ws + 7 * RB);       // mv gathered V bank
  bf16* o_bf   = (bf16*)(ws + 8 * RB);

  auto wptr = [&](int i) { return w_bf + (size_t)i * 1024 * 1024; };
  const int ELT_BLK = (NACT + 255) / 256;
  const int W_BLK   = (1024 * 1024 + 255) / 256;
  dim3 gemm_grid(96, 8);                     // M=12288, N=1024 in 128x128 tiles

  // Stage 0: conversions + residual
  cvt_f32_bf16_kernel<<<ELT_BLK, 256, 0, stream>>>(hs,  hs_bf,  NACT);
  cvt_f32_bf16_kernel<<<ELT_BLK, 256, 0, stream>>>(ref, ref_bf, NACT);
  for (int i = 0; i < 12; ++i)
    cvt_f32_bf16_kernel<<<W_BLK, 256, 0, stream>>>((const float*)d_in[2 + i],
                                                   wptr(i), 1024 * 1024);
  copy_f32_kernel<<<ELT_BLK, 256, 0, stream>>>(hs, out, NACT);

  // Dense branch runner (branch 1: self, branch 3: cross-ref)
  auto run_dense = [&](const bf16* kvsrc, int wq, int wk, int wv, int wo,
                       const float* bias) {
    gemm_bf16_kernel<0><<<gemm_grid, 256, 0, stream>>>(hs_bf, wptr(wq), q_bf, nullptr, nullptr);
    gemm_bf16_kernel<0><<<gemm_grid, 256, 0, stream>>>(kvsrc, wptr(wk), k_bf, nullptr, nullptr);
    gemm_bf16_kernel<0><<<gemm_grid, 256, 0, stream>>>(kvsrc, wptr(wv), v_bf, nullptr, nullptr);
    transpose_k_kernel<<<ELT_BLK, 256, 0, stream>>>(k_bf, kt_bf);
    attn_dense_kernel<<<dim3(8, 16, 12), 256, 0, stream>>>(q_bf, kt_bf, v_bf, o_bf);
    gemm_bf16_kernel<1><<<gemm_grid, 256, 0, stream>>>(o_bf, wptr(wo), nullptr, out, bias);
  };

  run_dense(hs_bf,  0, 1, 2, 3,  bout);       // self-attention
  run_dense(ref_bf, 8, 9, 10, 11, bout_ref);  // reference cross-attention

  // Multi-view row-wise branch
  gemm_bf16_kernel<0><<<gemm_grid, 256, 0, stream>>>(hs_bf, wptr(4), q_bf, nullptr, nullptr);
  gemm_bf16_kernel<0><<<gemm_grid, 256, 0, stream>>>(hs_bf, wptr(5), k_bf, nullptr, nullptr);
  gemm_bf16_kernel<0><<<gemm_grid, 256, 0, stream>>>(hs_bf, wptr(6), v_bf, nullptr, nullptr);
  gather_mv_kt_kernel<<<ELT_BLK, 256, 0, stream>>>(k_bf, kt_bf);
  gather_mv_v_kernel<<<ELT_BLK, 256, 0, stream>>>(v_bf, g_bf);
  attn_mv_kernel<<<1536, 256, 0, stream>>>(q_bf, kt_bf, g_bf, o_bf);
  gemm_bf16_kernel<1><<<gemm_grid, 256, 0, stream>>>(o_bf, wptr(7), nullptr, out, bout_mv);
}
